// LSTMValueEstimator_86861418595093
// MI455X (gfx1250) — matrix operand order repaired and checked by reading, not compiled, not run on hardware
//
#include <hip/hip_runtime.h>
#include <hip/hip_bf16.h>
#include <math.h>

// ---------------------------------------------------------------------------
// LSTMValueEstimator, fused for MI455X (gfx1250, wave32, WMMA).
//
// Kernel 1 (mlp_gates): per wave, 16 samples (one WMMA M-tile) are pushed
//   through 360->64->32->16->8 MLP + target concat + w_ih projection, all
//   with v_wmma_f32_16x16x32_f16. Weights live in LDS as f16, [n][k] layout.
//   Output: gates_in + b_ih + b_hh, f32, [T*B, 16] in workspace (8 MB).
// Kernel 2 (lstm_scan): 1 thread per batch element, sequential over T.
//
// Roofline: lidar = 188.7 MB read dominates -> ~8.1 us @ 23.3 TB/s.
// 6.8 GFLOP on f16 WMMA (f32 accum) is negligible vs that floor.
// ---------------------------------------------------------------------------

typedef __attribute__((ext_vector_type(16))) _Float16 v16h;
typedef __attribute__((ext_vector_type(8)))  _Float16 v8h;
typedef __attribute__((ext_vector_type(8)))  float    v8f;

#define T_STEPS 256
#define BATCH   512
#define LBEAMS  360
#define NSAMP   (T_STEPS * BATCH)   // 131072
#define NTILES  (NSAMP / 16)        // 8192 M-tiles
#define WAVES   4
#define NTHREADS (WAVES * 32)

#define W0_K   384                  // 360 padded up to 12 chunks of K=32
#define ACT_K  64                   // per-sample activation row stride (halves)

// LDS element counts
#define W0H_ELEMS (64 * W0_K)
#define W1H_ELEMS (32 * 64)
#define W2H_ELEMS (16 * 32)
#define W3H_ELEMS (16 * 32)
#define WIH_ELEMS (16 * 32)
#define SMEM_BYTES ((W0H_ELEMS + W1H_ELEMS + W2H_ELEMS + W3H_ELEMS + WIH_ELEMS) * 2 \
                    + (64 + 32 + 16 + 16 + 16) * 4 \
                    + WAVES * 16 * ACT_K * 2)

__device__ __forceinline__ v8h zero_v8h() {
  v8h z;
#pragma unroll
  for (int i = 0; i < 8; ++i) z[i] = (_Float16)0.f;
  return z;
}

// Convert two float4s (8 contiguous floats) to 8 packed halves.
__device__ __forceinline__ v8h cvt8(float4 x0, float4 x1) {
  v8h r;
  r[0] = (_Float16)x0.x; r[1] = (_Float16)x0.y;
  r[2] = (_Float16)x0.z; r[3] = (_Float16)x0.w;
  r[4] = (_Float16)x1.x; r[5] = (_Float16)x1.y;
  r[6] = (_Float16)x1.z; r[7] = (_Float16)x1.w;
  return r;
}

__device__ __forceinline__ v16h combine(v8h lo, v8h hi) {
  return __builtin_shufflevector(lo, hi, 0, 1, 2, 3, 4, 5, 6, 7,
                                 8, 9, 10, 11, 12, 13, 14, 15);
}

// --- A fragment (16-bit A 16x32 layout) ------------------------------------
// Lane owns row M=lane&15; per lane the fragment is one contiguous 16-element
// run: k = kbase + 16*khalf + {0..15}.
__device__ __forceinline__ v16h load_a_lds(const _Float16* __restrict__ row,
                                           int kbase, int khalf) {
  v8h lo = *(const v8h*)(row + kbase + 16 * khalf);
  v8h hi = *(const v8h*)(row + kbase + 16 * khalf + 8);
  return combine(lo, hi);
}

__device__ __forceinline__ v16h load_a_f32(const float* __restrict__ row,
                                           int kbase, int khalf) {
  const float* p = row + kbase + 16 * khalf;
  float4 x0 = ((const float4*)p)[0];
  float4 x1 = ((const float4*)p)[1];
  float4 y0 = ((const float4*)(p + 8))[0];
  float4 y1 = ((const float4*)(p + 8))[1];
  return combine(cvt8(x0, x1), cvt8(y0, y1));
}

// --- B fragment (16-bit B 32x16 layout) ------------------------------------
// Lane owns column N=lane&15; this code puts K=0..15 in lanes 0-15, K=16..31 in lanes 16-31 (same order as A).
// With weights stored [n][Kpad] the 16 halves are contiguous (32 B).
__device__ __forceinline__ v16h load_b_lds(const _Float16* __restrict__ wcol) {
  return *(const v16h*)wcol;
}

__global__ void __launch_bounds__(NTHREADS)
mlp_gates_kernel(const float* __restrict__ lidar,   // [T,B,360]
                 const float* __restrict__ target,  // [T,B,2]
                 const float* __restrict__ w0, const float* __restrict__ b0,
                 const float* __restrict__ w1, const float* __restrict__ b1,
                 const float* __restrict__ w2, const float* __restrict__ b2,
                 const float* __restrict__ w3, const float* __restrict__ b3,
                 const float* __restrict__ w_ih,
                 const float* __restrict__ b_ih, const float* __restrict__ b_hh,
                 float* __restrict__ gbuf)          // [T*B, 16]
{
  extern __shared__ char smem[];
  _Float16* w0h = (_Float16*)smem;              // [64][384]
  _Float16* w1h = w0h + W0H_ELEMS;              // [32][64]
  _Float16* w2h = w1h + W1H_ELEMS;              // [16][32]
  _Float16* w3h = w2h + W2H_ELEMS;              // [16][32] (rows>=8, k>=16 zero)
  _Float16* wih = w3h + W3H_ELEMS;              // [16][32] (k>=10 zero)
  float* bias0 = (float*)(wih + WIH_ELEMS);     // 64
  float* bias1 = bias0 + 64;                    // 32
  float* bias2 = bias1 + 32;                    // 16
  float* bias3 = bias2 + 16;                    // 16 (8 valid)
  float* biasg = bias3 + 16;                    // 16 (= b_ih + b_hh)
  _Float16* act = (_Float16*)(biasg + 16);      // [WAVES][16][ACT_K]

  const int tid = threadIdx.x;

  // ---- stage weights (f16, zero-padded) + biases into LDS -----------------
  for (int i = tid; i < W0H_ELEMS; i += NTHREADS) {
    int n = i / W0_K, k = i % W0_K;
    w0h[i] = (k < LBEAMS) ? (_Float16)w0[n * LBEAMS + k] : (_Float16)0.f;
  }
  for (int i = tid; i < W1H_ELEMS; i += NTHREADS) w1h[i] = (_Float16)w1[i];
  for (int i = tid; i < W2H_ELEMS; i += NTHREADS) w2h[i] = (_Float16)w2[i];
  for (int i = tid; i < W3H_ELEMS; i += NTHREADS) {
    int n = i / 32, k = i % 32;
    w3h[i] = (n < 8 && k < 16) ? (_Float16)w3[n * 16 + k] : (_Float16)0.f;
  }
  for (int i = tid; i < WIH_ELEMS; i += NTHREADS) {
    int n = i / 32, k = i % 32;
    wih[i] = (k < 10) ? (_Float16)w_ih[n * 10 + k] : (_Float16)0.f;
  }
  for (int i = tid; i < 64; i += NTHREADS) bias0[i] = b0[i];
  for (int i = tid; i < 32; i += NTHREADS) bias1[i] = b1[i];
  for (int i = tid; i < 16; i += NTHREADS) bias2[i] = b2[i];
  for (int i = tid; i < 16; i += NTHREADS) bias3[i] = (i < 8) ? b3[i] : 0.f;
  for (int i = tid; i < 16; i += NTHREADS) biasg[i] = b_ih[i] + b_hh[i];
  __syncthreads();

  const int lane   = tid & 31;
  const int wave   = tid >> 5;
  const int mrow   = lane & 15;        // A-fragment row owned by this lane
  const int khalf  = lane >> 4;        // A K-group select
  const int ncol   = lane & 15;        // B/C column owned by this lane
  const int mhalf8 = (lane >> 4) * 8;  // C-fragment row offset
  _Float16* myact = act + wave * 16 * ACT_K;
  const _Float16* myrow = myact + mrow * ACT_K;        // A-side view
  _Float16* crow = myact + mhalf8 * ACT_K + ncol;      // C-side view

  for (int tile = blockIdx.x * WAVES + wave; tile < NTILES;
       tile += gridDim.x * WAVES) {
    const int s0 = tile * 16;
    const float* arow = lidar + (size_t)(s0 + mrow) * LBEAMS;

    // prefetch next tile's lidar rows (global_prefetch_b8)
    int nxt = tile + gridDim.x * WAVES;
    if (nxt < NTILES)
      __builtin_prefetch(lidar + (size_t)(nxt * 16 + mrow) * LBEAMS, 0, 0);

    // ---- layer0: [16 x 360] x [360 x 64], K padded to 384 -----------------
    v8f acc0[4] = {};
#pragma unroll 1
    for (int kc = 0; kc < 11; ++kc) {          // full chunks: K 0..351
      v16h a = load_a_f32(arow, kc * 32, khalf);
#pragma unroll
      for (int nt = 0; nt < 4; ++nt) {
        v16h b = load_b_lds(w0h + (nt * 16 + ncol) * W0_K + kc * 32 + 16 * khalf);
        acc0[nt] = __builtin_amdgcn_wmma_f32_16x16x32_f16(
            false, a, false, b, (short)0, acc0[nt], false, false);
      }
    }
    {                                          // ragged tail: K 352..383
      v8h lo = zero_v8h();
      v8h hi = zero_v8h();
      if (khalf == 0) {                        // only K 352..359 are real
        const float* p = arow + 352;
        lo = cvt8(((const float4*)p)[0], ((const float4*)p)[1]);
      }
      v16h a = combine(lo, hi);
#pragma unroll
      for (int nt = 0; nt < 4; ++nt) {
        v16h b = load_b_lds(w0h + (nt * 16 + ncol) * W0_K + 11 * 32 + 16 * khalf);
        acc0[nt] = __builtin_amdgcn_wmma_f32_16x16x32_f16(
            false, a, false, b, (short)0, acc0[nt], false, false);
      }
    }
#pragma unroll
    for (int nt = 0; nt < 4; ++nt) {
      float bb = bias0[nt * 16 + ncol];
#pragma unroll
      for (int r = 0; r < 8; ++r)
        crow[r * ACT_K + nt * 16] = (_Float16)fmaxf(acc0[nt][r] + bb, 0.f);
    }

    // ---- layer1: [16 x 64] x [64 x 32] ------------------------------------
    v8f acc1[2] = {};
#pragma unroll
    for (int kc = 0; kc < 2; ++kc) {
      v16h a = load_a_lds(myrow, kc * 32, khalf);
#pragma unroll
      for (int nt = 0; nt < 2; ++nt) {
        v16h b = load_b_lds(w1h + (nt * 16 + ncol) * 64 + kc * 32 + 16 * khalf);
        acc1[nt] = __builtin_amdgcn_wmma_f32_16x16x32_f16(
            false, a, false, b, (short)0, acc1[nt], false, false);
      }
    }
#pragma unroll
    for (int nt = 0; nt < 2; ++nt) {
      float bb = bias1[nt * 16 + ncol];
#pragma unroll
      for (int r = 0; r < 8; ++r)
        crow[r * ACT_K + nt * 16] = (_Float16)fmaxf(acc1[nt][r] + bb, 0.f);
    }

    // ---- layer2: [16 x 32] x [32 x 16] ------------------------------------
    {
      v16h a = load_a_lds(myrow, 0, khalf);
      v16h b = load_b_lds(w2h + ncol * 32 + 16 * khalf);
      v8f acc2 = {};
      acc2 = __builtin_amdgcn_wmma_f32_16x16x32_f16(
          false, a, false, b, (short)0, acc2, false, false);
      float bb = bias2[ncol];
#pragma unroll
      for (int r = 0; r < 8; ++r) {
        crow[r * ACT_K]      = (_Float16)fmaxf(acc2[r] + bb, 0.f);
        crow[r * ACT_K + 16] = (_Float16)0.f;          // zero-pad K 16..31
      }
    }

    // ---- layer3: [16 x 16(pad32)] x [16(pad32) x 8(pad16)] + target concat
    {
      v16h a = load_a_lds(myrow, 0, khalf);
      v16h b = load_b_lds(w3h + ncol * 32 + 16 * khalf);
      v8f acc3 = {};
      acc3 = __builtin_amdgcn_wmma_f32_16x16x32_f16(
          false, a, false, b, (short)0, acc3, false, false);
      const float* tp = target + (size_t)(s0 + mhalf8) * 2;
      float bb = bias3[ncol];
#pragma unroll
      for (int r = 0; r < 8; ++r) {
        _Float16 val;
        if (ncol < 8)       val = (_Float16)fmaxf(acc3[r] + bb, 0.f);
        else if (ncol == 8) val = (_Float16)tp[r * 2 + 0];
        else if (ncol == 9) val = (_Float16)tp[r * 2 + 1];
        else                val = (_Float16)0.f;
        crow[r * ACT_K] = val;             // [16..31] already zero from layer2
      }
    }

    // ---- gates: [16 x 10(pad32)] x [10(pad32) x 16] -> workspace ----------
    {
      v16h a = load_a_lds(myrow, 0, khalf);
      v16h b = load_b_lds(wih + ncol * 32 + 16 * khalf);
      v8f accg = {};
      accg = __builtin_amdgcn_wmma_f32_16x16x32_f16(
          false, a, false, b, (short)0, accg, false, false);
      float bb = biasg[ncol];
      float* gout = gbuf + (size_t)(s0 + mhalf8) * 16 + ncol;
#pragma unroll
      for (int r = 0; r < 8; ++r)
        gout[r * 16] = accg[r] + bb;       // constant 64B strides -> imm offs
    }
  }
}

// ---------------------------------------------------------------------------
// Kernel 2: LSTM scan. 1 thread per batch element, T sequential, HID=4.
// gates are pre-biased (b_ih + b_hh folded in).
// ---------------------------------------------------------------------------
__device__ __forceinline__ float sigmoidf(float x) {
  return 1.f / (1.f + __expf(-x));
}

__global__ void lstm_scan_kernel(const float* __restrict__ gbuf,  // [T*B,16]
                                 const float* __restrict__ w_hh,  // [16,4]
                                 float* __restrict__ out)         // [B,4]
{
  __shared__ float whh[64];
  if (threadIdx.x < 64) whh[threadIdx.x] = w_hh[threadIdx.x];
  __syncthreads();

  int b = blockIdx.x * blockDim.x + threadIdx.x;
  if (b >= BATCH) return;

  float h[4] = {0.f, 0.f, 0.f, 0.f};
  float c[4] = {0.f, 0.f, 0.f, 0.f};
  const float* g = gbuf + (size_t)b * 16;
  for (int t = 0; t < T_STEPS; ++t, g += (size_t)BATCH * 16) {
    float gv[16];
#pragma unroll
    for (int j = 0; j < 16; ++j) {
      float s = g[j];
#pragma unroll
      for (int q = 0; q < 4; ++q) s += whh[j * 4 + q] * h[q];
      gv[j] = s;
    }
#pragma unroll
    for (int k = 0; k < 4; ++k) {
      float ig = sigmoidf(gv[k]);
      float fg = sigmoidf(gv[4 + k]);
      float gc = tanhf(gv[8 + k]);
      float og = sigmoidf(gv[12 + k]);
      c[k] = fg * c[k] + ig * gc;
      h[k] = og * tanhf(c[k]);
    }
  }
#pragma unroll
  for (int k = 0; k < 4; ++k) out[(size_t)b * 4 + k] = h[k];
}

// ---------------------------------------------------------------------------
extern "C" void kernel_launch(void* const* d_in, const int* in_sizes, int n_in,
                              void* d_out, int out_size, void* d_ws, size_t ws_size,
                              hipStream_t stream) {
  const float* lidar  = (const float*)d_in[0];
  const float* target = (const float*)d_in[1];
  const float* w0   = (const float*)d_in[2];
  const float* b0   = (const float*)d_in[3];
  const float* w1   = (const float*)d_in[4];
  const float* b1   = (const float*)d_in[5];
  const float* w2   = (const float*)d_in[6];
  const float* b2   = (const float*)d_in[7];
  const float* w3   = (const float*)d_in[8];
  const float* b3   = (const float*)d_in[9];
  const float* w_ih = (const float*)d_in[10];
  const float* w_hh = (const float*)d_in[11];
  const float* b_ih = (const float*)d_in[12];
  const float* b_hh = (const float*)d_in[13];

  float* gbuf = (float*)d_ws;  // needs NSAMP*16*4 = 8 MB

  mlp_gates_kernel<<<dim3(512), dim3(NTHREADS), SMEM_BYTES, stream>>>(
      lidar, target, w0, b0, w1, b1, w2, b2, w3, b3, w_ih, b_ih, b_hh, gbuf);

  lstm_scan_kernel<<<dim3((BATCH + 127) / 128), dim3(128), 0, stream>>>(
      gbuf, w_hh, (float*)d_out);
}
